// PhysicsNetwork_38362647887998
// MI455X (gfx1250) — matrix operand order repaired and checked by reading, not compiled, not run on hardware
//
#include <hip/hip_runtime.h>

typedef __bf16 bf16;
typedef __attribute__((ext_vector_type(16))) __bf16 v16bf;
typedef __attribute__((ext_vector_type(8)))  __bf16 v8bf;
typedef __attribute__((ext_vector_type(8)))  float  v8f;

// ---- constants -------------------------------------------------------------
#define KOBJ 16
#define REP 128
#define ASZ 6
#define AENC 64
#define BSK 32768           // 2048 * 16 flat batch rows

// repacked-weight offsets (bf16 elements) inside d_ws.  pe_W1 is split into
// top (rows 0..127) and bottom (rows 128..255) halves, each 128->256.
constexpr int OFF_LE_W1   = 0;
constexpr int OFF_LE_W2   = 16384;
constexpr int OFF_AE_W1   = 32768;
constexpr int OFF_AE_W2   = 36864;
constexpr int OFF_AEF_W1  = 45056;
constexpr int OFF_AEF_W2  = 69632;
constexpr int OFF_AAT_W1  = 86016;
constexpr int OFF_AAT_W2  = 110592;
constexpr int OFF_PE_W1T  = 112640;
constexpr int OFF_PE_W1B  = 145408;
constexpr int OFF_PE_W2   = 178176;
constexpr int OFF_IE_W1   = 210944;
constexpr int OFF_IE_W2   = 227328;
constexpr int OFF_IA_W1   = 235520;
constexpr int OFF_IA_W2   = 251904;
constexpr int OFF_FM_W1   = 253952;
constexpr int OFF_FM_W2   = 278528;
constexpr int WTOT        = 294912;

constexpr long ENC_FLAT_OFF = WTOT;                         // BSK*128 bf16
constexpr long U_OFF        = ENC_FLAT_OFF + (long)BSK*128; // BSK*256 bf16
constexpr long V_OFF        = U_OFF        + (long)BSK*256; // BSK*256 bf16
constexpr long TOTEFF_OFF   = V_OFF        + (long)BSK*256; // BSK*64  bf16

// ---- fragment loader + activation ------------------------------------------
__device__ __forceinline__ v16bf load_a_frag(const bf16* Alds, int sa, int kt,
                                             int row, int ksel)
{
    const bf16* pa = Alds + row * sa + kt * 32 + ksel;
    v8bf lo = *(const v8bf*)pa;          // K = base .. base+7
    v8bf hi = *(const v8bf*)(pa + 16);   // K = base+16 .. base+23
    v16bf a;
#pragma unroll
    for (int t = 0; t < 8; ++t) { a[t] = lo[t]; a[t + 8] = hi[t]; }
    return a;
}

template<int ACT>
__device__ __forceinline__ float act_apply(float v)
{
    if (ACT == 1) v = (v > 0.0f) ? v : (__expf(v) - 1.0f);
    if (ACT == 2) v = 1.0f / (1.0f + __expf(-v));
    return v;
}

// ---- single-tile wave-level WMMA MLP layer ---------------------------------
// ACT: 0 = identity, 1 = elu, 2 = sigmoid.  All shape/dest flags compile-time.
template<int ACT, int KT, int FOUT, bool HB, bool OB, bool OB2, bool OF, bool OG>
__device__ __forceinline__ void wave_mlp(
    const bf16* __restrict__ Alds, int sa,
    const bf16* __restrict__ Wfrag, const float* __restrict__ bias,
    bf16* out_bf, int so, int col_off,              // LDS bf16 out      (OB)
    bf16* out_bf2, int so2,                         // global bf16 out   (OB2)
    float* outf, int sf,                            // LDS f32 out       (OF)
    float* __restrict__ out_gl, int sg,             // global f32 out    (OG)
    int lane)
{
    constexpr int NT = (FOUT + 15) >> 4;
    constexpr bool FULL = (FOUT & 15) == 0;
    const int row  = lane & 15;
    const int ksel = (lane >> 4) << 3;
    const int mb   = (lane >> 4) << 3;

    for (int nt = 0; nt < NT; ++nt) {
        v8f acc = {};
        for (int kt = 0; kt < KT; ++kt) {
            v16bf a = load_a_frag(Alds, sa, kt, row, ksel);
            v16bf b = *(const v16bf*)(Wfrag + (((kt * NT + nt) * 32 + lane) << 4));
            acc = __builtin_amdgcn_wmma_f32_16x16x32_bf16(
                false, a, false, b, (short)0, acc, false, false);
        }
        const int n  = nt * 16 + (lane & 15);
        const bool nv = FULL || (n < FOUT);
        const float bv = (HB && nv) ? bias[n] : 0.0f;
#pragma unroll
        for (int r = 0; r < 8; ++r) {
            const float v = act_apply<ACT>(acc[r] + bv);
            if (nv) {
                const int m = mb + r;
                if (OB)  out_bf [m * so  + col_off + n] = (bf16)v;
                if (OB2) out_bf2[m * so2 + n]           = (bf16)v;
                if (OF)  outf   [m * sf  + n]           = v;
                if (OG)  out_gl [m * sg  + n]           = v;
            }
        }
    }
}

// ---- dual-tile layer: two A tiles share every B fragment -------------------
template<int ACT, int KT, int FOUT, bool OB, bool OF>
__device__ __forceinline__ void wave_mlp_dual(
    const bf16* __restrict__ A0, const bf16* __restrict__ A1, int sa,
    const bf16* __restrict__ Wfrag, const float* __restrict__ bias,
    bf16* o0, bf16* o1, int so,                     // LDS bf16 outs (OB)
    float* f0, float* f1, int sf,                   // LDS f32 outs  (OF)
    int lane)
{
    constexpr int NT = (FOUT + 15) >> 4;
    constexpr bool FULL = (FOUT & 15) == 0;
    const int row  = lane & 15;
    const int ksel = (lane >> 4) << 3;
    const int mb   = (lane >> 4) << 3;

    for (int nt = 0; nt < NT; ++nt) {
        v8f acc0 = {}, acc1 = {};
        for (int kt = 0; kt < KT; ++kt) {
            v16bf a0 = load_a_frag(A0, sa, kt, row, ksel);
            v16bf a1 = load_a_frag(A1, sa, kt, row, ksel);
            v16bf b = *(const v16bf*)(Wfrag + (((kt * NT + nt) * 32 + lane) << 4));
            acc0 = __builtin_amdgcn_wmma_f32_16x16x32_bf16(
                false, a0, false, b, (short)0, acc0, false, false);
            acc1 = __builtin_amdgcn_wmma_f32_16x16x32_bf16(
                false, a1, false, b, (short)0, acc1, false, false);
        }
        const int n  = nt * 16 + (lane & 15);
        const bool nv = FULL || (n < FOUT);
        const float bv = nv ? bias[n] : 0.0f;
#pragma unroll
        for (int r = 0; r < 8; ++r) {
            const float v0 = act_apply<ACT>(acc0[r] + bv);
            const float v1 = act_apply<ACT>(acc1[r] + bv);
            if (nv) {
                const int m = mb + r;
                if (OB) { o0[m * so + n] = (bf16)v0; o1[m * so + n] = (bf16)v1; }
                if (OF) { f0[m * sf + n] = v0;       f1[m * sf + n] = v1; }
            }
        }
    }
}

// ---- weight repack: f32 row-major -> bf16 WMMA B-fragment order ------------
struct PrepArgs { const float* src[17]; };

__global__ __launch_bounds__(256) void prep_weights(PrepArgs P, bf16* __restrict__ wsW)
{
    const int FIN[17]  = {128,128,  6,128, 192,128, 192,128, 128,128, 256, 128,128, 128,128, 192,128};
    const int FOUT[17] = {128,128,128, 64, 128,128, 128,  1, 256,256, 128, 128, 64, 128,  1, 128,128};
    const int OFF[17]  = {OFF_LE_W1, OFF_LE_W2, OFF_AE_W1, OFF_AE_W2,
                          OFF_AEF_W1, OFF_AEF_W2, OFF_AAT_W1, OFF_AAT_W2,
                          OFF_PE_W1T, OFF_PE_W1B, OFF_PE_W2,
                          OFF_IE_W1, OFF_IE_W2, OFF_IA_W1, OFF_IA_W2,
                          OFF_FM_W1, OFF_FM_W2};
    const int m = blockIdx.x;
    const float* __restrict__ src = P.src[m];
    const int fin = FIN[m], fout = FOUT[m];
    const int ktiles = (fin + 31) >> 5, ntiles = (fout + 15) >> 4;
    const int total = ktiles * ntiles * 512;
    bf16* __restrict__ dst = wsW + OFF[m];
    for (int idx = threadIdx.x; idx < total; idx += 256) {
        const int e    = idx & 15;
        const int lane = (idx >> 4) & 31;
        const int f    = idx >> 9;
        const int nt   = f % ntiles, kt = f / ntiles;
        const int n = nt * 16 + (lane & 15);
        const int k = kt * 32 + ((lane >> 4) << 3) + (e & 7) + ((e >> 3) << 4);   // lanes 16-31: K+8
        const float v = (k < fin && n < fout) ? src[k * fout + n] : 0.0f;
        dst[idx] = (bf16)v;
    }
}

// ---- stage 1: encoder + per-row pe-layer-1 partials U/V --------------------
struct EncArgs {
    const float *lambda1, *actions;
    const float *le_b1, *le_b2, *ae_b1, *ae_b2, *aef_b1, *aef_b2, *aat_b1, *aat_b2;
    const float *pe_b1;
    const bf16* wsW;
    bf16 *enc_flat, *U, *V;
};

__global__ __launch_bounds__(64) void enc_kernel(EncArgs P)
{
    __shared__ __align__(16) bf16 sA[2][16 * 200];
    __shared__ __align__(16) bf16 sB[2][16 * 136];
    __shared__ __align__(16) bf16 sC[2][16 * 136];
    __shared__ float sAtt[2][16];

    const int w = threadIdx.x >> 5, lane = threadIdx.x & 31;
    const int m0 = (blockIdx.x * 2 + w) * 16;
    bf16* A = sA[w]; bf16* B = sB[w]; bf16* C = sC[w]; float* att = sAtt[w];

    // lambda1 tile -> A (stride 136)
    for (int idx = lane; idx < 16 * 128; idx += 32) {
        const int r = idx >> 7, c = idx & 127;
        A[r * 136 + c] = (bf16)P.lambda1[(long)(m0 + r) * REP + c];
    }
    // le: 128 ->(elu) 128 ->(elu) 128 ; enc_flat also to global bf16
    wave_mlp<1,4,128, true, true,false,false,false>(A, 136, P.wsW + OFF_LE_W1, P.le_b1,
                B, 136, 0, nullptr, 0, nullptr, 0, nullptr, 0, lane);
    wave_mlp<1,4,128, true, true,true,false,false>(B, 136, P.wsW + OFF_LE_W2, P.le_b2,
                C, 136, 0, P.enc_flat + (long)m0 * REP, REP, nullptr, 0, nullptr, 0, lane);

    // actions tile -> A (stride 40, padded to 32 cols)
    for (int idx = lane; idx < 16 * 32; idx += 32) {
        const int r = idx >> 5, c = idx & 31;
        A[r * 40 + c] = (bf16)((c < ASZ) ? P.actions[(long)(m0 + r) * ASZ + c] : 0.0f);
    }
    // ae: 6(pad32) ->(elu) 128 ->(elu) 64, a_enc into la cols [128,192)
    wave_mlp<1,1,128, true, true,false,false,false>(A, 40, P.wsW + OFF_AE_W1, P.ae_b1,
                B, 136, 0, nullptr, 0, nullptr, 0, nullptr, 0, lane);
    wave_mlp<1,4,64, true, true,false,false,false>(B, 136, P.wsW + OFF_AE_W2, P.ae_b2,
                A, 200, 128, nullptr, 0, nullptr, 0, nullptr, 0, lane);
    for (int idx = lane; idx < 16 * 128; idx += 32) {   // la cols [0,128)
        const int r = idx >> 7, c = idx & 127;
        A[r * 200 + c] = C[r * 136 + c];
    }
    // aef: 192 ->(elu) 128 ->(elu) 128  => a_eff in C
    wave_mlp<1,6,128, true, true,false,false,false>(A, 200, P.wsW + OFF_AEF_W1, P.aef_b1,
                B, 136, 0, nullptr, 0, nullptr, 0, nullptr, 0, lane);
    wave_mlp<1,4,128, true, true,false,false,false>(B, 136, P.wsW + OFF_AEF_W2, P.aef_b2,
                C, 136, 0, nullptr, 0, nullptr, 0, nullptr, 0, lane);
    // aat: 192 ->(elu) 128 ->(sigmoid) 1 => att[16]
    wave_mlp<1,6,128, true, true,false,false,false>(A, 200, P.wsW + OFF_AAT_W1, P.aat_b1,
                B, 136, 0, nullptr, 0, nullptr, 0, nullptr, 0, lane);
    wave_mlp<2,4,1, true, false,false,true,false>(B, 136, P.wsW + OFF_AAT_W2, P.aat_b2,
                nullptr, 0, 0, nullptr, 0, att, 1, nullptr, 0, lane);
    // enc tile = a_eff * att -> B (LDS bf16)
    for (int idx = lane; idx < 16 * 128; idx += 32) {
        const int r = idx >> 7, c = idx & 127;
        B[r * 136 + c] = (bf16)((float)C[r * 136 + c] * att[r]);
    }
    // pe layer-1 per-row partials: U = enc@W1_top + pe_b1 ; V = enc@W1_bot
    wave_mlp<0,4,256, true, false,true,false,false>(B, 136, P.wsW + OFF_PE_W1T, P.pe_b1,
                nullptr, 0, 0, P.U + (long)m0 * 256, 256, nullptr, 0, nullptr, 0, lane);
    wave_mlp<0,4,256, false, false,true,false,false>(B, 136, P.wsW + OFF_PE_W1B, nullptr,
                nullptr, 0, 0, P.V + (long)m0 * 256, 256, nullptr, 0, nullptr, 0, lane);
}

// ---- stage 2: pairwise, two groups per wave (shared B fragments) -----------
struct PairArgs {
    const float *pe_b2, *ie_b1, *ie_b2, *ia_b1, *ia_b2;
    const bf16* wsW;
    const bf16 *U, *V;
    bf16* toteff;
};

__global__ __launch_bounds__(32) void pair_kernel(PairArgs P)
{
    __shared__ __align__(16) bf16  sH [2][16 * 264]; // hidden (256-wide), reused 128-wide
    __shared__ __align__(16) bf16  sPi[2][16 * 136];
    __shared__ __align__(16) float sEff[2][16 * 64];
    __shared__ float sAtt[2][16];

    const int lane = threadIdx.x & 31;
    const int g0 = blockIdx.x * 2, g1 = g0 + 1;      // same 16-row base always
    const int base = g0 & ~15;
    const int i0 = g0 & 15, i1 = g1 & 15;

    // hidden tiles: h[j] = elu( U[g] + V[base+j] )   (V rows loaded once)
    const unsigned int* __restrict__ Uw = (const unsigned int*)P.U;   // 128 u32/row
    const unsigned int* __restrict__ Vw = (const unsigned int*)P.V;
    for (int idx = lane; idx < 16 * 128; idx += 32) {
        const int j = idx >> 7, c = idx & 127;
        const unsigned int vv = Vw[(long)(base + j) * 128 + c];
        const unsigned int u0 = Uw[(long)g0 * 128 + c];
        const unsigned int u1 = Uw[(long)g1 * 128 + c];
        const float vlo = __uint_as_float(vv << 16);
        const float vhi = __uint_as_float(vv & 0xffff0000u);
        float a = act_apply<1>(__uint_as_float(u0 << 16) + vlo);
        float b = act_apply<1>(__uint_as_float(u0 & 0xffff0000u) + vhi);
        sH[0][j * 264 + 2 * c]     = (bf16)a;
        sH[0][j * 264 + 2 * c + 1] = (bf16)b;
        a = act_apply<1>(__uint_as_float(u1 << 16) + vlo);
        b = act_apply<1>(__uint_as_float(u1 & 0xffff0000u) + vhi);
        sH[1][j * 264 + 2 * c]     = (bf16)a;
        sH[1][j * 264 + 2 * c + 1] = (bf16)b;
    }
    // pe L2: 256 ->(elu) 128
    wave_mlp_dual<1,8,128, true,false>(sH[0], sH[1], 264, P.wsW + OFF_PE_W2, P.pe_b2,
                     sPi[0], sPi[1], 136, nullptr, nullptr, 0, lane);
    // ie: 128 ->(elu) 128 ->(elu) 64 (f32)
    wave_mlp_dual<1,4,128, true,false>(sPi[0], sPi[1], 136, P.wsW + OFF_IE_W1, P.ie_b1,
                     sH[0], sH[1], 136, nullptr, nullptr, 0, lane);
    wave_mlp_dual<1,4,64, false,true>(sH[0], sH[1], 136, P.wsW + OFF_IE_W2, P.ie_b2,
                     nullptr, nullptr, 0, sEff[0], sEff[1], 64, lane);
    // ia: 128 ->(elu) 128 ->(sigmoid) 1
    wave_mlp_dual<1,4,128, true,false>(sPi[0], sPi[1], 136, P.wsW + OFF_IA_W1, P.ia_b1,
                     sH[0], sH[1], 136, nullptr, nullptr, 0, lane);
    wave_mlp_dual<2,4,1, false,true>(sH[0], sH[1], 136, P.wsW + OFF_IA_W2, P.ia_b2,
                     nullptr, nullptr, 0, sAtt[0], sAtt[1], 1, lane);
    if (lane == 0) { sAtt[0][i0] = 0.0f; sAtt[1][i1] = 0.0f; }  // drop j == i
    // total_effect[g,:] = sum_j effect[j,:] * att[j]
#pragma unroll
    for (int grp = 0; grp < 2; ++grp) {
        const float* eff = sEff[grp];
        const float* att = sAtt[grp];
        const long g = g0 + grp;
        for (int c = lane; c < 64; c += 32) {
            float s = 0.0f;
#pragma unroll
            for (int j = 0; j < 16; ++j) s += eff[j * 64 + c] * att[j];
            P.toteff[g * AENC + c] = (bf16)s;
        }
    }
}

// ---- stage 3: final merge (fm) -> d_out f32 --------------------------------
struct FinArgs {
    const float *fm_b1, *fm_b2;
    const bf16* wsW;
    const bf16 *enc_flat, *toteff;
    float* out;
};

__global__ __launch_bounds__(64) void fin_kernel(FinArgs P)
{
    __shared__ __align__(16) bf16 sA[2][16 * 200];
    __shared__ __align__(16) bf16 sB[2][16 * 136];
    const int w = threadIdx.x >> 5, lane = threadIdx.x & 31;
    const int m0 = (blockIdx.x * 2 + w) * 16;
    bf16* A = sA[w]; bf16* B = sB[w];

    // la2 = [ enc_flat(128) | total_effect(64) ]  (u32 copies of bf16)
    const unsigned int* __restrict__ efw = (const unsigned int*)P.enc_flat; // 64 u32/row
    const unsigned int* __restrict__ tew = (const unsigned int*)P.toteff;   // 32 u32/row
    unsigned int* dst = (unsigned int*)A;                                    // 100 u32/row
    for (int idx = lane; idx < 16 * 96; idx += 32) {
        const int r = idx / 96, c = idx % 96;
        const unsigned int v = (c < 64) ? efw[(long)(m0 + r) * 64 + c]
                                        : tew[(long)(m0 + r) * 32 + (c - 64)];
        dst[r * 100 + c] = v;
    }
    // fm: 192 ->(elu) 128 ->(identity) 128 -> global f32
    wave_mlp<1,6,128, true, true,false,false,false>(A, 200, P.wsW + OFF_FM_W1, P.fm_b1,
                B, 136, 0, nullptr, 0, nullptr, 0, nullptr, 0, lane);
    wave_mlp<0,4,128, true, false,false,false,true>(B, 136, P.wsW + OFF_FM_W2, P.fm_b2,
                nullptr, 0, 0, nullptr, 0, nullptr, 0,
                P.out + (long)m0 * REP, REP, lane);
}

// ---- host launch -----------------------------------------------------------
extern "C" void kernel_launch(void* const* d_in, const int* in_sizes, int n_in,
                              void* d_out, int out_size, void* d_ws, size_t ws_size,
                              hipStream_t stream)
{
    (void)in_sizes; (void)n_in; (void)out_size; (void)ws_size;
    bf16* ws = (bf16*)d_ws;

    PrepArgs pa;
    const int widx[16] = {2, 4, 6, 8, 10, 12, 14, 16, 18, 20, 22, 24, 26, 28, 30, 32};
    const float* W[16];
    for (int i = 0; i < 16; ++i) W[i] = (const float*)d_in[widx[i]];
    for (int i = 0; i < 8; ++i) pa.src[i] = W[i];     // le/ae/aef/aat
    pa.src[8] = W[8];                                  // pe_W1 top (rows 0..127)
    pa.src[9] = W[8] + 128 * 256;                      // pe_W1 bottom (rows 128..255)
    for (int i = 9; i < 16; ++i) pa.src[i + 1] = W[i]; // pe_W2 .. fm_W2
    prep_weights<<<dim3(17), dim3(256), 0, stream>>>(pa, ws);

    EncArgs ea{ (const float*)d_in[0],  (const float*)d_in[1],
                (const float*)d_in[3],  (const float*)d_in[5],
                (const float*)d_in[7],  (const float*)d_in[9],
                (const float*)d_in[11], (const float*)d_in[13],
                (const float*)d_in[15], (const float*)d_in[17],
                (const float*)d_in[19],
                ws, ws + ENC_FLAT_OFF, ws + U_OFF, ws + V_OFF };
    enc_kernel<<<dim3(BSK / 32), dim3(64), 0, stream>>>(ea);

    PairArgs pb{ (const float*)d_in[21],
                 (const float*)d_in[23], (const float*)d_in[25],
                 (const float*)d_in[27], (const float*)d_in[29],
                 ws, ws + U_OFF, ws + V_OFF, ws + TOTEFF_OFF };
    pair_kernel<<<dim3(BSK / 2), dim3(32), 0, stream>>>(pb);   // 2 groups per wave

    FinArgs fc{ (const float*)d_in[31], (const float*)d_in[33],
                ws, ws + ENC_FLAT_OFF, ws + TOTEFF_OFF, (float*)d_out };
    fin_kernel<<<dim3(BSK / 32), dim3(64), 0, stream>>>(fc);
}
